// NGramAttention_42906723287566
// MI455X (gfx1250) — matrix operand order repaired and checked by reading, not compiled, not run on hardware
//
#include <hip/hip_runtime.h>
#include <math.h>

// ---------------------------------------------------------------------------
// NGramAttention on gfx1250 (MI455X), wave32 + WMMA.
//  - input->hidden GRU projections: v_wmma_f32_16x16x32_f16, f16 operands,
//    A-fragment reused across the 3 gate N-tiles, weights pre-padded to f16
//  - recurrent h@Whh.T: 3 WMMAs/step in one resident workgroup, h in LDS,
//    Gi loads software-pipelined one step ahead, no per-step barrier
//  - attention projection (N padded 75->80) also on WMMA
//  - conv / tanh / softmax-over-batch / pooling / dense head on VALU
// ---------------------------------------------------------------------------

typedef __attribute__((ext_vector_type(16))) _Float16 v16h;
typedef __attribute__((ext_vector_type(8)))  float    v8f;

#define T_STEPS 1024
#define BATCH   128
#define HID     16

__device__ __forceinline__ float sigmoidf_(float x) { return 1.0f / (1.0f + __expf(-x)); }

// ---- zero-pad + f32->f16 weight repack: Wp[Npad x Kpad] ----
__global__ void wpad16_kernel(const float* __restrict__ W, _Float16* __restrict__ Wp,
                              int N, int K, int Kpad, int total) {
  int idx = blockIdx.x * blockDim.x + threadIdx.x;
  if (idx >= total) return;
  int n = idx / Kpad, k = idx % Kpad;
  Wp[idx] = (n < N && k < K) ? (_Float16)W[n * K + k] : (_Float16)0.0f;
}

// ---- conv(k,1) + transpose(1,3,0,2) + K-pad to 96, writes [T*B, Kpad] f16 ----
__global__ void conv_pad_kernel(const float* __restrict__ inp, const float* __restrict__ cw,
                                const float* __restrict__ cb, _Float16* __restrict__ xpad,
                                int ksz, int Hk, int Kpad) {
  int idx = blockIdx.x * blockDim.x + threadIdx.x;
  int total = T_STEPS * BATCH * Kpad;
  if (idx >= total) return;
  int f  = idx % Kpad;
  int tb = idx / Kpad;
  int b  = tb % BATCH;
  int t  = tb / BATCH;
  int c = t >> 2, w = t & 3;            // t = channel*4 + width
  float v = 0.0f;
  if (f < Hk) {
    v = cb[c];
    for (int i = 0; i < ksz; ++i)
      v += cw[c * ksz + i] * inp[b * 300 + (f + i) * 4 + w];   // inputs [B,1,75,4]
  }
  xpad[idx] = (_Float16)v;
}

// ---- Gi = A @ Wih.T + bih : one wave per 16-row M tile, 3 N-tiles, K<=96 ----
__global__ void gemm_ih_wmma(const _Float16* __restrict__ A, const _Float16* __restrict__ Wp,
                             const float* __restrict__ bias, float* __restrict__ Gi,
                             int Kpad) {
  int lane = threadIdx.x;
  int mt = blockIdx.x;
  int row = mt * 16 + (lane & 15);      // A rows (lanes 16-31 repeat rows)
  int nnl = lane & 15;                  // B column within tile
  int kbA = (lane < 16) ? 0 : 16;       // A: K 0-15 vs 16-31
  int kbB = (lane < 16) ? 0 : 16;       // B: K 0-15 vs 16-31
  v8f a0 = {}, a1 = {}, a2 = {};
  for (int k0 = 0; k0 < Kpad; k0 += 32) {
    const _Float16* arow = A + (size_t)row * Kpad + k0;
    v16h af, b0, b1, b2;
#pragma unroll
    for (int e = 0; e < 8; ++e) {
      af[e]     = arow[kbA + e];
      af[8 + e] = arow[kbA + 8 + e];
    }
#pragma unroll
    for (int e = 0; e < 16; ++e) {
      int k = k0 + kbB + e;
      b0[e] = Wp[(0 * 16 + nnl) * Kpad + k];
      b1[e] = Wp[(1 * 16 + nnl) * Kpad + k];
      b2[e] = Wp[(2 * 16 + nnl) * Kpad + k];
    }
    a0 = __builtin_amdgcn_wmma_f32_16x16x32_f16(false, af, false, b0, (short)0, a0, false, false);
    a1 = __builtin_amdgcn_wmma_f32_16x16x32_f16(false, af, false, b1, (short)0, a1, false, false);
    a2 = __builtin_amdgcn_wmma_f32_16x16x32_f16(false, af, false, b2, (short)0, a2, false, false);
  }
  int mb = (lane < 16) ? 0 : 8;         // C/D: M = v + 8*(lane>=16)
#pragma unroll
  for (int v = 0; v < 8; ++v) {
    size_t base = (size_t)(mt * 16 + mb + v) * 48;
    Gi[base + nnl]      = a0[v] + bias[nnl];
    Gi[base + 16 + nnl] = a1[v] + bias[16 + nnl];
    Gi[base + 32 + nnl] = a2[v] + bias[32 + nnl];
  }
}

// ---- sequential GRU scan: 8 waves, h in LDS, 3 WMMAs/step, Gi pipelined ----
__global__ void __launch_bounds__(256) gru_scan_wmma(
    const float* __restrict__ Gi, const float* __restrict__ Whh,
    const float* __restrict__ bhh, float* __restrict__ out,
    _Float16* __restrict__ out16, int dir) {
  __shared__ float hlds[BATCH][HID];
  int tid = threadIdx.x;
  int lane = tid & 31;
  int m = tid >> 5;                     // wave id = exclusive 16-row batch tile
  for (int i = tid; i < BATCH * HID; i += 256) ((float*)hlds)[i] = 0.0f;
  __syncthreads();

  // Whh B-fragments (3 gate tiles), K=16 real + 16 zero pad
  v16h Bf0, Bf1, Bf2;
#pragma unroll
  for (int e = 0; e < 16; ++e) {
    int c = lane & 15;
    Bf0[e] = (lane < 16) ? (_Float16)Whh[(0 * 16 + c) * HID + e] : (_Float16)0.0f;
    Bf1[e] = (lane < 16) ? (_Float16)Whh[(1 * 16 + c) * HID + e] : (_Float16)0.0f;
    Bf2[e] = (lane < 16) ? (_Float16)Whh[(2 * 16 + c) * HID + e] : (_Float16)0.0f;
  }
  int j = lane & 15;
  float bh0 = bhh[j], bh1 = bhh[16 + j], bh2 = bhh[32 + j];
  int mb   = (lane < 16) ? 0 : 8;
  int kbA  = (lane < 16) ? 0 : 8;
  int arow = m * 16 + (lane & 15);

  // software-pipeline: preload Gi for step 0
  float gr[8], gz[8], gn[8];
  {
    int tt0 = dir ? (T_STEPS - 1) : 0;
#pragma unroll
    for (int v = 0; v < 8; ++v) {
      size_t base = ((size_t)tt0 * BATCH + m * 16 + mb + v) * 48;
      gr[v] = Gi[base + j]; gz[v] = Gi[base + 16 + j]; gn[v] = Gi[base + 32 + j];
    }
  }

  for (int step = 0; step < T_STEPS; ++step) {
    int tt = dir ? (T_STEPS - 1 - step) : step;
    v16h af;
#pragma unroll
    for (int e = 0; e < 8; ++e) {
      af[e]     = (lane < 16) ? (_Float16)hlds[arow][e] : (_Float16)0.0f;
      af[8 + e] = (lane < 16) ? (_Float16)hlds[arow][8 + e] : (_Float16)0.0f;
    }
    v8f c0 = {}, c1 = {}, c2 = {};
    c0 = __builtin_amdgcn_wmma_f32_16x16x32_f16(false, af, false, Bf0, (short)0, c0, false, false);
    c1 = __builtin_amdgcn_wmma_f32_16x16x32_f16(false, af, false, Bf1, (short)0, c1, false, false);
    c2 = __builtin_amdgcn_wmma_f32_16x16x32_f16(false, af, false, Bf2, (short)0, c2, false, false);

    // issue next step's Gi loads while WMMAs are in flight
    float ngr[8] = {0}, ngz[8] = {0}, ngn[8] = {0};
    bool more = (step + 1 < T_STEPS);
    if (more) {
      int ttn = dir ? (T_STEPS - 2 - step) : (step + 1);
#pragma unroll
      for (int v = 0; v < 8; ++v) {
        size_t base = ((size_t)ttn * BATCH + m * 16 + mb + v) * 48;
        ngr[v] = Gi[base + j]; ngz[v] = Gi[base + 16 + j]; ngn[v] = Gi[base + 32 + j];
      }
    }

    float hn[8];
#pragma unroll
    for (int v = 0; v < 8; ++v) {
      int b = m * 16 + mb + v;
      float hp = hlds[b][j];
      float r = sigmoidf_(gr[v] + c0[v] + bh0);
      float z = sigmoidf_(gz[v] + c1[v] + bh1);
      float n = tanhf(gn[v] + r * (c2[v] + bh2));
      hn[v] = (1.0f - z) * n + z * hp;
    }
#pragma unroll
    for (int v = 0; v < 8; ++v) {
      int b = m * 16 + mb + v;
      size_t o = ((size_t)tt * BATCH + b) * 32 + dir * 16 + j;
      hlds[b][j] = hn[v];
      out[o] = hn[v];
      out16[o] = (_Float16)hn[v];
      gr[v] = ngr[v]; gz[v] = ngz[v]; gn[v] = ngn[v];
    }
  }
}

// ---- attention projection: pre = H16 @ attw.T + attb, N padded to 80 ----
__global__ void gemm_att_wmma(const _Float16* __restrict__ H16, const _Float16* __restrict__ Wp,
                              const float* __restrict__ bias, float* __restrict__ pre,
                              int Hk) {
  int lane = threadIdx.x;
  int mt = blockIdx.x, nt = blockIdx.y;
  int row = mt * 16 + (lane & 15);
  int nn  = nt * 16 + (lane & 15);
  int kbA = (lane < 16) ? 0 : 16;
  int kbB = (lane < 16) ? 0 : 16;
  const _Float16* arow = H16 + (size_t)row * 32;
  v16h af, bf;
#pragma unroll
  for (int e = 0; e < 8; ++e) {
    af[e]     = arow[kbA + e];
    af[8 + e] = arow[kbA + 8 + e];
  }
#pragma unroll
  for (int e = 0; e < 16; ++e) bf[e] = Wp[nn * 32 + kbB + e];
  v8f acc = {};
  acc = __builtin_amdgcn_wmma_f32_16x16x32_f16(false, af, false, bf, (short)0, acc, false, false);
  float bi = (nn < Hk) ? bias[nn] : 0.0f;
  int mb = (lane < 16) ? 0 : 8;
#pragma unroll
  for (int v = 0; v < 8; ++v)
    pre[(size_t)(mt * 16 + mb + v) * 80 + nn] = acc[v] + bi;
}

// ---- a = tanh(pre) in place; s = a @ ctx ----
__global__ void att_s_kernel(float* __restrict__ pre, const float* __restrict__ ctx,
                             float* __restrict__ sbuf, int Hk) {
  int tb = blockIdx.x * blockDim.x + threadIdx.x;
  if (tb >= T_STEPS * BATCH) return;
  float* pr = pre + (size_t)tb * 80;
  float s = 0.0f;
  for (int d = 0; d < Hk; ++d) {
    float a = tanhf(pr[d]);
    pr[d] = a;
    s += a * ctx[d];
  }
  sbuf[tb] = s;
}

// ---- softmax over the BATCH axis for each t (faithful torch dim=1 quirk) ----
__global__ void softmax_b_kernel(const float* __restrict__ s, float* __restrict__ alpha) {
  __shared__ float red[BATCH];
  int t = blockIdx.x, b = threadIdx.x;
  float v = s[t * BATCH + b];
  red[b] = v;
  __syncthreads();
  for (int off = 64; off > 0; off >>= 1) {
    if (b < off) red[b] = fmaxf(red[b], red[b + off]);
    __syncthreads();
  }
  float mx = red[0];
  __syncthreads();
  float e = __expf(v - mx);
  red[b] = e;
  __syncthreads();
  for (int off = 64; off > 0; off >>= 1) {
    if (b < off) red[b] += red[b + off];
    __syncthreads();
  }
  alpha[t * BATCH + b] = e / red[0];
}

// ---- p[d,b] = sum_t a[t,b,d] * alpha[t,b] ----
__global__ void att_p_kernel(const float* __restrict__ abuf, const float* __restrict__ alpha,
                             float* __restrict__ pout, int Hk) {
  int idx = blockIdx.x * blockDim.x + threadIdx.x;
  if (idx >= Hk * BATCH) return;
  int b = idx % BATCH, d = idx / BATCH;
  float acc = 0.0f;
  for (int t = 0; t < T_STEPS; ++t)
    acc += abuf[((size_t)t * BATCH + b) * 80 + d] * alpha[t * BATCH + b];
  pout[d * BATCH + b] = acc;
}

// ---- dense1(222->128)+ReLU then dense2(128->2); one block per batch row ----
__global__ void dense_kernel(const float* __restrict__ x, const float* __restrict__ w1,
                             const float* __restrict__ b1, const float* __restrict__ w2,
                             const float* __restrict__ b2, float* __restrict__ out) {
  __shared__ float hm[128];
  int b = blockIdx.x, o = threadIdx.x;
  float acc = b1[o];
  for (int d = 0; d < 222; ++d) acc += x[d * BATCH + b] * w1[o * 222 + d];
  hm[o] = fmaxf(acc, 0.0f);
  __syncthreads();
  if (o < 2) {
    float a2 = b2[o];
    for (int k = 0; k < 128; ++k) a2 += hm[k] * w2[o * 128 + k];
    out[b * 2 + o] = a2;
  }
}

// ---------------------------------------------------------------------------
extern "C" void kernel_launch(void* const* d_in, const int* in_sizes, int n_in,
                              void* d_out, int out_size, void* d_ws, size_t ws_size,
                              hipStream_t stream) {
  (void)in_sizes; (void)n_in; (void)out_size; (void)ws_size;
  float* ws = (float*)d_ws;
  // workspace layout (float units); ~170 MB total, fits in the 192 MB L2
  float*     giF  = ws;                       //  6,291,456  [131072,48] f32
  float*     giB  = giF  + 6291456;           //  6,291,456
  float*     outA = giB  + 6291456;           //  4,194,304  [131072,32] f32
  float*     outB = outA + 4194304;           //  4,194,304
  _Float16*  h16A = (_Float16*)(outB + 4194304);   // [131072,32] f16 (2,097,152 fl)
  _Float16*  h16B = h16A + 4194304;                // (2,097,152 fl)
  _Float16*  x16  = h16B + 4194304;                // [131072,96] f16 (6,291,456 fl)
  float*     pre  = (float*)(x16 + 12582912);      // 10,485,760  [131072,80] f32
  _Float16*  wp0  = (_Float16*)(pre + 10485760);   // 48*96 f16
  _Float16*  wp1  = wp0 + 4608;
  _Float16*  wpa  = wp1 + 4608;                    // 80*32 f16
  float*     sbuf = (float*)(wpa + 2560);          // 131,072
  float*     albf = sbuf + 131072;                 // 131,072
  float*     pbuf = albf + 131072;                 // 28,416

  const float* inputs = (const float*)d_in[0];

  // d_in flat order: inputs, (conv_w, conv_b) x3, gru1(24), gru2(16), gru3(8),
  // (att_w, att_b) x3, ctx x3, dense1_w/b, dense2_w/b
  struct Br { int cwi, cbi, gbase, nlayers, ksz, awi, abi, ci, poff; };
  const Br brs[3] = {
    {1, 2,  7, 3, 1, 55, 56, 61,   0},
    {3, 4, 31, 2, 2, 57, 58, 62,  75},
    {5, 6, 47, 1, 3, 59, 60, 63, 149},
  };

  for (int bi = 0; bi < 3; ++bi) {
    const Br& br = brs[bi];
    int Hk = 76 - br.ksz;

    { // conv + reshape + zero-pad K to 96 (f16)
      int tot = T_STEPS * BATCH * 96;
      conv_pad_kernel<<<(tot + 255) / 256, 256, 0, stream>>>(
          inputs, (const float*)d_in[br.cwi], (const float*)d_in[br.cbi], x16, br.ksz, Hk, 96);
    }

    const _Float16* lin = x16;
    int K = Hk, Kpad = 96;
    float* lout = outA;
    _Float16* lout16 = h16A;
    for (int l = 0; l < br.nlayers; ++l) {
      for (int dir = 0; dir < 2; ++dir) {
        int pb = br.gbase + (l * 2 + dir) * 4;
        _Float16* wp = dir ? wp1 : wp0;
        int wtot = 48 * Kpad;
        wpad16_kernel<<<(wtot + 255) / 256, 256, 0, stream>>>(
            (const float*)d_in[pb + 0], wp, 48, K, Kpad, wtot);
        gemm_ih_wmma<<<8192, 32, 0, stream>>>(
            lin, wp, (const float*)d_in[pb + 2], dir ? giB : giF, Kpad);
      }
      for (int dir = 0; dir < 2; ++dir) {
        int pb = br.gbase + (l * 2 + dir) * 4;
        gru_scan_wmma<<<1, 256, 0, stream>>>(
            dir ? giB : giF, (const float*)d_in[pb + 1], (const float*)d_in[pb + 3],
            lout, lout16, dir);
      }
      lin = lout16; K = 32; Kpad = 32;
      lout   = (lout == outA) ? outB : outA;
      lout16 = (lout16 == h16A) ? h16B : h16A;
    }

    // attention: WMMA projection (N 75->80 pad), tanh+s, softmax(batch), pool
    {
      int wtot = 80 * 32;
      wpad16_kernel<<<(wtot + 255) / 256, 256, 0, stream>>>(
          (const float*)d_in[br.awi], wpa, Hk, 32, 32, wtot);
      gemm_att_wmma<<<dim3(8192, 5), 32, 0, stream>>>(
          lin, wpa, (const float*)d_in[br.abi], pre, Hk);
      att_s_kernel<<<512, 256, 0, stream>>>(pre, (const float*)d_in[br.ci], sbuf, Hk);
      softmax_b_kernel<<<T_STEPS, BATCH, 0, stream>>>(sbuf, albf);
      att_p_kernel<<<(Hk * BATCH + 255) / 256, 256, 0, stream>>>(
          pre, albf, pbuf + br.poff * BATCH, Hk);
    }
  }

  dense_kernel<<<BATCH, 128, 0, stream>>>(
      pbuf, (const float*)d_in[64], (const float*)d_in[65],
      (const float*)d_in[66], (const float*)d_in[67], (float*)d_out);
}
